// Transformer_77412490543795
// MI455X (gfx1250) — matrix operand order repaired and checked by reading, not compiled, not run on hardware
//
#include <hip/hip_runtime.h>
#include <hip/hip_bf16.h>

// ---------------- problem constants (from reference) ----------------
#define DD    256
#define DIH   1024
#define NLAY  4
#define HNUM  4
#define BNUM  4
#define LLEN  2048
#define SEGC  334
#define HALFC 166
#define NAC   8
#define NTC   10

typedef _Float16 h16;
typedef __attribute__((ext_vector_type(16))) _Float16 v16h;
typedef __attribute__((ext_vector_type(8)))  _Float16 v8h;
typedef __attribute__((ext_vector_type(8)))  float    v8f;
typedef __attribute__((ext_vector_type(4)))  unsigned int u32x4;
typedef __attribute__((ext_vector_type(8)))  int      i32x8;
typedef __attribute__((ext_vector_type(4)))  int      i32x4;

// ---- Tensor Data Mover availability (toolchain-dependent arity) ----
#ifndef HAVE_TDM
#define HAVE_TDM 0
#endif
#if defined(__has_builtin)
#if __has_builtin(__builtin_amdgcn_tensor_load_to_lds)
#undef HAVE_TDM
#define HAVE_TDM 1
#endif
#endif

__device__ __forceinline__ v8f wmma_f16(v16h a, v16h b, v8f c) {
  // D = A(16x32 f16) * B(32x16 f16) + C(16x16 f32)
  return __builtin_amdgcn_wmma_f32_16x16x32_f16(false, a, false, b,
                                                (short)0, c, false, false);
}

// contiguous 16-halves LDS read; compiler merges to 2x ds_load_b128
__device__ __forceinline__ v16h ld_frag(const h16* p) {
  v16h f;
  #pragma unroll
  for (int e = 0; e < 16; ++e) f[e] = p[e];
  return f;
}

__device__ __forceinline__ float gelu_tanh(float x) {
  const float k0 = 0.7978845608028654f, k1 = 0.044715f;
  float x3 = x * x * x;
  return 0.5f * x * (1.f + tanhf(k0 * (x + k1 * x3)));
}

#define EPI_GELU 1

// ---------------- WMMA tiled GEMM: C[M,N] = act(A[M,K]*W[K,N] + bias) -------
// BM=128, BN=64, BK=32. 256 threads = 8 waves, wave grid 4(m) x 2(n),
// each wave: 2x2 of 16x16 f32 accumulators -> 4 v_wmma per K-step.
// A tile: f32 in LDS, row-major [row][k] — filled by TDM (tensor_load_to_lds)
//         when available, else cooperative float4 staging. Fragments convert
//         f32->f16 with v_cvt_pk on the fly.
// B tile: f16 in LDS, K-major [col][k] (contiguous B-frag reads).
#define GBM 128
#define GBN 64
#define GBK 32

__global__ __launch_bounds__(256)
void k_gemm(const float* __restrict__ A, const float* __restrict__ W,
            const float* __restrict__ bias, float* __restrict__ C,
            int M, int N, int K, int flags)
{
  __shared__ float sAf32[2][GBM * GBK];  // 16 KB x2
  __shared__ h16   sB[2][GBN * GBK];     //  4 KB x2

  const int tid  = threadIdx.x;
  const int wave = tid >> 5, lane = tid & 31;
  const int hi = lane >> 4, ln = lane & 15;
  const int wm = (wave & 3) * 32;    // wave row offset inside block tile
  const int wn = (wave >> 2) * 32;   // wave col offset inside block tile
  const int bm = blockIdx.x * GBM;
  const int bn = blockIdx.y * GBN;

  v8f acc[2][2] = {};

#if HAVE_TDM
  // Issue one TDM tile load (global f32 tile -> LDS), wave 0 only.
  auto tdm_issueA = [&](int buf, int k0) {
    unsigned lds = (unsigned)(size_t)&sAf32[buf][0];
    unsigned long long ga =
        (unsigned long long)(size_t)(A + (size_t)bm * K + k0);
    int rows = M - bm; if (rows > GBM) rows = GBM; if (rows < 0) rows = 0;
    u32x4 g0;
    g0[0] = 1u;                                   // count=1, user mode
    g0[1] = lds;                                  // lds_addr (bytes)
    g0[2] = (unsigned)(ga & 0xffffffffu);         // global_addr[31:0]
    g0[3] = (unsigned)((ga >> 32) & 0x1ffffffu)   // global_addr[56:32]
          | 0x80000000u;                          // type=2 ("image")
    i32x8 g1;
    g1[0] = (int)(2u << 16);                      // data_size=4B
    g1[1] = (int)(((unsigned)K & 0xffffu) << 16); // tensor_dim0[15:0]
    g1[2] = (int)(((unsigned)K >> 16)             // tensor_dim0[31:16]
          | (((unsigned)rows & 0xffffu) << 16));  // tensor_dim1[15:0]
    g1[3] = (int)(((unsigned)rows >> 16)          // tensor_dim1[31:16]
          | ((unsigned)GBK << 16));               // tile_dim0 = 32
    g1[4] = (int)GBM;                             // tile_dim1 = 128
    g1[5] = (int)K;                               // tensor_dim0_stride[31:0]
    g1[6] = 0;
    g1[7] = 0;
    i32x4 z4 = {0, 0, 0, 0};
#if __clang_major__ >= 23
    i32x8 z8 = {0, 0, 0, 0, 0, 0, 0, 0};
    __builtin_amdgcn_tensor_load_to_lds(g0, g1, z4, z4, z8, 0);
#else
    __builtin_amdgcn_tensor_load_to_lds(g0, g1, z4, z4, 0);
#endif
  };
#endif
  // Fallback / non-TDM cooperative staging of the f32 A tile.
  auto stageA = [&](int buf, int k0) {
    #pragma unroll
    for (int i = 0; i < 16; i += 4) {
      int idx = tid * 16 + i;
      int r = idx >> 5, c = idx & 31;            // 128 rows x 32 cols
      int gr = bm + r;
      float4 f = make_float4(0.f, 0.f, 0.f, 0.f);
      if (gr < M) f = *(const float4*)(A + (size_t)gr * K + k0 + c);
      *(float4*)&sAf32[buf][idx] = f;
    }
  };
  // K-major staging of B: thread covers one column, 8 consecutive k values.
  auto loadB = [&](int buf, int k0) {
    int c  = tid & 63;            // column within tile
    int kg = (tid >> 6) * 8;      // k-group
    const float* src = W + (size_t)(k0 + kg) * N + bn + c;
    __builtin_prefetch(src + (size_t)GBK * N, 0, 1);  // -> global_prefetch_b8
    v8h pk;
    #pragma unroll
    for (int i = 0; i < 8; ++i) pk[i] = (h16)src[(size_t)i * N];
    *(v8h*)&sB[buf][c * GBK + kg] = pk;
  };
  // A-frag: per-lane two contiguous 8-float runs, converted to f16
  auto fragA = [&](int buf, int mt) -> v16h {
    int row = wm + mt * 16 + ln;
    const float* base = &sAf32[buf][row * GBK];
    v16h f;
    #pragma unroll
    for (int e = 0; e < 8; ++e) f[e] = (h16)base[(hi << 4) + e];
    #pragma unroll
    for (int e = 0; e < 8; ++e) f[8 + e] = (h16)base[(hi << 4) + 8 + e];
    return f;
  };
  // B-frag: k = (hi<<4)|e, n = ln -> contiguous 16 halves
  auto fragB = [&](int buf, int nt) -> v16h {
    int col = wn + nt * 16 + ln;
    return ld_frag(&sB[buf][col * GBK + (hi << 4)]);
  };

#if HAVE_TDM
  if (wave == 0) tdm_issueA(0, 0);
#else
  stageA(0, 0);
#endif
  loadB(0, 0);
#if HAVE_TDM
  if (wave == 0) __builtin_amdgcn_s_wait_tensorcnt(0);
#endif
  __syncthreads();

  const int nk = K >> 5;
  for (int kk = 0; kk < nk; ++kk) {
    int cur = kk & 1;
    if (kk + 1 < nk) {
#if HAVE_TDM
      if (wave == 0) tdm_issueA(cur ^ 1, (kk + 1) << 5);  // overlap DMA w/ WMMA
#else
      stageA(cur ^ 1, (kk + 1) << 5);
#endif
      loadB(cur ^ 1, (kk + 1) << 5);
    }
    v16h a0 = fragA(cur, 0), a1 = fragA(cur, 1);
    v16h b0 = fragB(cur, 0), b1 = fragB(cur, 1);
    acc[0][0] = wmma_f16(a0, b0, acc[0][0]);
    acc[0][1] = wmma_f16(a0, b1, acc[0][1]);
    acc[1][0] = wmma_f16(a1, b0, acc[1][0]);
    acc[1][1] = wmma_f16(a1, b1, acc[1][1]);
#if HAVE_TDM
    if (kk + 1 < nk && wave == 0) __builtin_amdgcn_s_wait_tensorcnt(0);
#endif
    __syncthreads();
  }

  // C/D layout: row = r + 8*hi, col = ln
  #pragma unroll
  for (int mt = 0; mt < 2; ++mt)
    #pragma unroll
    for (int nt = 0; nt < 2; ++nt)
      #pragma unroll
      for (int r = 0; r < 8; ++r) {
        int row = bm + wm + mt * 16 + r + hi * 8;
        int col = bn + wn + nt * 16 + ln;
        if (row < M) {
          float v = acc[mt][nt][r] + (bias ? bias[col] : 0.f);
          if (flags & EPI_GELU) v = gelu_tanh(v);
          C[(size_t)row * N + col] = v;
        }
      }
}

// ---------------- flash attention (WMMA, online softmax) ----------------
// q,k,v,o: [B*Sq, 256] with col = h*64+d. Block = 8 waves over 128 query rows
// of one (b,h); K/V tiles staged cooperatively in LDS per 32-column block.
__global__ __launch_bounds__(256)
void k_flash(const float* __restrict__ Qp, const float* __restrict__ Kp,
             const float* __restrict__ Vp, float* __restrict__ Op,
             const int* __restrict__ types, int Lfull, int use_mask, int Sq)
{
  __shared__ h16 sK[32 * 64];      // [col(32)][k(64)]   (K-major)
  __shared__ h16 sV[64 * 32];      // [dim(64)][row(32)] (transposed)
  __shared__ h16 sP[8][16 * 32];   // per-wave P relayout scratch
  const int tid = threadIdx.x;
  const int wave = tid >> 5, lane = tid & 31;
  const int hi = lane >> 4, ln = lane & 15;
  const int b = blockIdx.z, h = blockIdx.y;
  const int row0 = blockIdx.x * 128 + wave * 16;

  const float qscale = 0.125f;  // 1/sqrt(64)

  v16h qf[2];
  {
    int qr = row0 + ln; if (qr >= Sq) qr = Sq - 1;
    const float* qrow = Qp + (size_t)(b * Sq + qr) * DD + h * 64;
    #pragma unroll
    for (int f = 0; f < 2; ++f)
      #pragma unroll
      for (int e = 0; e < 16; ++e) {
        int k = (f << 5) | (hi << 4) | e;
        qf[f][e] = (h16)(qrow[k] * qscale);
      }
  }

  float mrow[8], lrow[8];
  #pragma unroll
  for (int r = 0; r < 8; ++r) { mrow[r] = -1e30f; lrow[r] = 0.f; }
  v8f oacc[4] = {};

  for (int j0 = 0; j0 < Sq; j0 += 32) {
    // ---- cooperative staging ----
    {
      // K: thread -> (row kr, 8-wide k chunk). 2x float4 coalesced reads.
      int kr = tid >> 3, kc = (tid & 7) * 8;
      int col = j0 + kr;
      v8h pk = {};
      if (col < Sq) {
        const float* src = Kp + (size_t)(b * Sq + col) * DD + h * 64 + kc;
        float4 f0 = *(const float4*)src;
        float4 f1 = *(const float4*)(src + 4);
        pk[0] = (h16)f0.x; pk[1] = (h16)f0.y; pk[2] = (h16)f0.z; pk[3] = (h16)f0.w;
        pk[4] = (h16)f1.x; pk[5] = (h16)f1.y; pk[6] = (h16)f1.z; pk[7] = (h16)f1.w;
      }
      *(v8h*)&sK[kr * 64 + kc] = pk;
      // V (transposed): thread -> (dim c, 8 rows). Coalesced b32 reads.
      int c = tid & 63, r0v = (tid >> 6) * 8;
      v8h pv;
      #pragma unroll
      for (int i = 0; i < 8; ++i) {
        int vr = j0 + r0v + i;
        pv[i] = (vr < Sq) ? (h16)Vp[(size_t)(b * Sq + vr) * DD + h * 64 + c]
                          : (h16)0.f;
      }
      *(v8h*)&sV[c * 32 + r0v] = pv;
    }
    __syncthreads();

    // ---- S = Q K^T for 16 x 32 block ----
    v8f sacc[2] = {};
    #pragma unroll
    for (int ct = 0; ct < 2; ++ct) {
      #pragma unroll
      for (int f = 0; f < 2; ++f) {
        v16h bf = ld_frag(&sK[(ct * 16 + ln) * 64 + (f << 5) + (hi << 4)]);
        sacc[ct] = wmma_f16(qf[f], bf, sacc[ct]);
      }
    }
    // ---- mask + online softmax ----
    float pbuf[2][8];
    #pragma unroll
    for (int ct = 0; ct < 2; ++ct) {
      int col = j0 + ct * 16 + ln;
      bool oob = (col >= Sq);
      bool kpad = use_mask && !oob && (types[b * Lfull + col] == 0);
      #pragma unroll
      for (int r = 0; r < 8; ++r) {
        int rowg = row0 + r + hi * 8;
        float s = sacc[ct][r];
        if (oob) s = -1e30f;
        else if (use_mask && (col > rowg || kpad)) s = -1e9f;
        pbuf[ct][r] = s;
      }
    }
    #pragma unroll
    for (int r = 0; r < 8; ++r) {
      float mx = fmaxf(pbuf[0][r], pbuf[1][r]);
      #pragma unroll
      for (int m = 1; m < 16; m <<= 1) mx = fmaxf(mx, __shfl_xor(mx, m, 32));
      float mnew = fmaxf(mrow[r], mx);
      float corr = __expf(mrow[r] - mnew);
      float p0 = __expf(pbuf[0][r] - mnew);
      float p1 = __expf(pbuf[1][r] - mnew);
      pbuf[0][r] = p0; pbuf[1][r] = p1;
      float ss = p0 + p1;
      #pragma unroll
      for (int m = 1; m < 16; m <<= 1) ss += __shfl_xor(ss, m, 32);
      lrow[r] = lrow[r] * corr + ss;
      mrow[r] = mnew;
      #pragma unroll
      for (int nt = 0; nt < 4; ++nt) oacc[nt][r] *= corr;
    }
    // ---- relayout P (C-layout -> A-layout) through per-wave LDS ----
    #pragma unroll
    for (int ct = 0; ct < 2; ++ct)
      #pragma unroll
      for (int r = 0; r < 8; ++r)
        sP[wave][(r + hi * 8) * 32 + ct * 16 + ln] = (h16)pbuf[ct][r];
    v16h pf;
    {
      const h16* base = &sP[wave][ln * 32];
      #pragma unroll
      for (int e = 0; e < 8; ++e) pf[e] = base[(hi << 4) + e];
      #pragma unroll
      for (int e = 0; e < 8; ++e) pf[8 + e] = base[(hi << 4) + 8 + e];
    }
    // ---- O += P V ----
    #pragma unroll
    for (int nt = 0; nt < 4; ++nt) {
      v16h vf = ld_frag(&sV[(nt * 16 + ln) * 32 + (hi << 4)]);
      oacc[nt] = wmma_f16(pf, vf, oacc[nt]);
    }
    __syncthreads();   // protect sK/sV before next staging pass
  }

  #pragma unroll
  for (int nt = 0; nt < 4; ++nt)
    #pragma unroll
    for (int r = 0; r < 8; ++r) {
      int rowg = row0 + r + hi * 8;
      if (rowg < Sq) {
        float den = lrow[r] > 0.f ? lrow[r] : 1.f;
        Op[(size_t)(b * Sq + rowg) * DD + h * 64 + nt * 16 + ln] = oacc[nt][r] / den;
      }
    }
}

// ---------------- layernorm with residual:  Y = LN(P + R)*g + c, * npm -----
__global__ __launch_bounds__(256)
void k_ln(const float* __restrict__ P, const float* __restrict__ R,
          const float* __restrict__ g, const float* __restrict__ c,
          const float* __restrict__ npm, int Lfull, int off,
          float* __restrict__ Y, int Srows)
{
  __shared__ float red[256];
  int row = blockIdx.x;
  int b = row / Srows, s = row - b * Srows;
  int t = threadIdx.x;
  size_t base = (size_t)row * DD;
  float v = P[base + t] + R[base + t];
  red[t] = v; __syncthreads();
  for (int st = 128; st > 0; st >>= 1) { if (t < st) red[t] += red[t + st]; __syncthreads(); }
  float mean = red[0] * (1.f / DD); __syncthreads();
  float d = v - mean;
  red[t] = d * d; __syncthreads();
  for (int st = 128; st > 0; st >>= 1) { if (t < st) red[t] += red[t + st]; __syncthreads(); }
  float var = red[0] * (1.f / DD);
  float npmv = npm[b * Lfull + off + s];
  Y[base + t] = (d * rsqrtf(var + 1e-6f) * g[t] + c[t]) * npmv;
}

// ---------------- small elementwise / helper kernels ----------------
__global__ void k_sincos(const float* __restrict__ tsrc, const float* __restrict__ npm,
                         float* __restrict__ out, int Srows, int Lfull, int off, int total)
{
  int idx = blockIdx.x * blockDim.x + threadIdx.x;
  if (idx >= total) return;
  int d = idx & (DD - 1);
  int rs = idx >> 8;
  int b = rs / Srows, s = rs - b * Srows;
  float tv = tsrc[b * Lfull + off + s];
  float r = tv * __powf(10000.f, -2.f * (float)(d >> 1) * (1.f / DD));
  float val = ((d & 1) == 0) ? __sinf(r) : __cosf(r);
  out[idx] = val * npm[b * Lfull + off + s];
}

__global__ void k_embed(const int* __restrict__ types, const float* __restrict__ emb,
                        float* __restrict__ out, int Srows, int Lfull, int off, int total)
{
  int idx = blockIdx.x * blockDim.x + threadIdx.x;
  if (idx >= total) return;
  int d = idx & (DD - 1);
  int rs = idx >> 8;
  int b = rs / Srows, s = rs - b * Srows;
  int tt = types[b * Lfull + off + s];
  out[idx] = emb[tt * DD + d];
}

__global__ void k_add(float* __restrict__ dst, const float* __restrict__ src, int n)
{
  int i = blockIdx.x * blockDim.x + threadIdx.x;
  if (i < n) dst[i] += src[i];
}

__global__ void k_zero(float* __restrict__ p, int n)
{
  int i = blockIdx.x * blockDim.x + threadIdx.x;
  if (i < n) p[i] = 0.f;
}

// attention probabilities for (b=0, h=H-1), rows [SEG-HALF,SEG), full-row softmax,
// keep last HALF columns.  q,k are the last decoder layer's projections.
__global__ __launch_bounds__(256)
void k_attblock(const float* __restrict__ Qp, const float* __restrict__ Kp,
                float* __restrict__ att)
{
  __shared__ float srow[SEGC];
  __shared__ float red[256];
  int r = blockIdx.x;                 // 0..HALF-1
  int qrow = SEGC - HALFC + r;        // b = 0
  int t = threadIdx.x;
  const float* q = Qp + (size_t)qrow * DD + (HNUM - 1) * 64;
  for (int m = t; m < SEGC; m += 256) {
    const float* k = Kp + (size_t)m * DD + (HNUM - 1) * 64;
    float acc = 0.f;
    for (int d = 0; d < 64; ++d) acc += q[d] * k[d];
    srow[m] = acc * 0.125f;
  }
  __syncthreads();
  float mx = -1e30f;
  for (int m = t; m < SEGC; m += 256) mx = fmaxf(mx, srow[m]);
  red[t] = mx; __syncthreads();
  for (int st = 128; st > 0; st >>= 1) { if (t < st) red[t] = fmaxf(red[t], red[t + st]); __syncthreads(); }
  mx = red[0]; __syncthreads();
  float sm = 0.f;
  for (int m = t; m < SEGC; m += 256) { float e = __expf(srow[m] - mx); srow[m] = e; sm += e; }
  red[t] = sm; __syncthreads();
  for (int st = 128; st > 0; st >>= 1) { if (t < st) red[t] += red[t + st]; __syncthreads(); }
  sm = red[0]; __syncthreads();
  for (int cc = t; cc < HALFC; cc += 256) att[r * HALFC + cc] = srow[SEGC - HALFC + cc] / sm;
}

__global__ void k_graph(const float* __restrict__ att, const int* __restrict__ tpred,
                        int Lfull, float* __restrict__ G)
{
  int idx = blockIdx.x * blockDim.x + threadIdx.x;
  if (idx >= HALFC * HALFC) return;
  int i = idx / HALFC, j = idx - i * HALFC;
  int a = tpred[(Lfull - SEGC) + i] - 1;
  int bb = tpred[(Lfull - SEGC) + j] - 1;
  if (a >= 0 && a < NTC && bb >= 0 && bb < NTC)
    atomicAdd(&G[a * NTC + bb], att[idx]);
}

__global__ __launch_bounds__(128)
void k_retrieve(const float* __restrict__ G, const float* __restrict__ Xs,
                int* __restrict__ ind)
{
  __shared__ float z[NTC * NTC], outv[NTC * NTC], probs[NAC];
  __shared__ float red[128];
  __shared__ int eq[NAC];
  int t = threadIdx.x;
  float gv = (t < NTC * NTC) ? G[t] : 0.f;
  red[t] = gv; __syncthreads();
  for (int st = 64; st > 0; st >>= 1) { if (t < st) red[t] += red[t + st]; __syncthreads(); }
  float mean = red[0] * (1.f / (NTC * NTC));
  if (t < NTC * NTC) z[t] = (gv < mean) ? 0.f : gv;
  __syncthreads();
  if (t < NAC) {
    float acc = 0.f;
    for (int c2 = 0; c2 < NTC * NTC; ++c2) acc += Xs[t * NTC * NTC + c2] * z[c2];
    probs[t] = 8.f * acc;  // BETA
  }
  __syncthreads();
  if (t == 0) {
    float mx = probs[0];
    for (int r2 = 1; r2 < NAC; ++r2) mx = fmaxf(mx, probs[r2]);
    float sm = 0.f;
    for (int r2 = 0; r2 < NAC; ++r2) { probs[r2] = __expf(probs[r2] - mx); sm += probs[r2]; }
    for (int r2 = 0; r2 < NAC; ++r2) probs[r2] /= sm;
  }
  __syncthreads();
  if (t < NTC * NTC) {
    float acc = 0.f;
    for (int r2 = 0; r2 < NAC; ++r2) acc += probs[r2] * Xs[r2 * NTC * NTC + t];
    outv[t] = acc;
  }
  __syncthreads();
  if (t < NAC) {
    int e = 1;
    for (int c2 = 0; c2 < NTC * NTC; ++c2)
      if (Xs[t * NTC * NTC + c2] != outv[c2]) { e = 0; break; }
    eq[t] = e;
  }
  __syncthreads();
  if (t == 0) {
    int id = 0;
    for (int r2 = 0; r2 < NAC; ++r2) if (!eq[r2]) id = r2;
    *ind = id;
  }
}

__global__ void k_concat(const float* __restrict__ X, const float* __restrict__ encs,
                         const int* __restrict__ ind, float* __restrict__ OP, int total)
{
  int idx = blockIdx.x * blockDim.x + threadIdx.x;
  if (idx >= total) return;
  int d = idx & (DD - 1);
  int rs = idx >> 8;
  int b = rs / LLEN, l = rs - b * LLEN;
  if (l < LLEN - HALFC) OP[idx] = X[idx];
  else {
    int ii = *ind;
    OP[idx] = encs[(((size_t)ii * BNUM + b) * HALFC + (l - (LLEN - HALFC))) * DD + d];
  }
}

// ---------------- host orchestration ----------------
struct LayerP {
  const float *wq, *bq, *wk, *bk, *wv, *bv, *wo, *bo;
  const float *g1, *c1, *w1, *b1, *w2, *b2, *g2, *c2;
};

static inline void launch_gemm(hipStream_t st, const float* A, const float* W,
                               const float* bias, float* C, int M, int N, int K,
                               int flags)
{
  dim3 g((M + GBM - 1) / GBM, N / GBN);
  k_gemm<<<g, dim3(256), 0, st>>>(A, W, bias, C, M, N, K, flags);
}

struct Ctx {
  float *Qb, *Kb, *Vb, *Ob, *P1, *Hb;
};

static void run_layer(hipStream_t st, float* A, int Sq, int use_mask,
                      const int* types, const float* npm, int off,
                      const LayerP& p, const Ctx& w)
{
  const int M = BNUM * Sq;
  launch_gemm(st, A, p.wq, p.bq, w.Qb, M, DD, DD, 0);
  launch_gemm(st, A, p.wk, p.bk, w.Kb, M, DD, DD, 0);
  launch_gemm(st, A, p.wv, p.bv, w.Vb, M, DD, DD, 0);
  dim3 fg((Sq + 127) / 128, HNUM, BNUM);
  k_flash<<<fg, dim3(256), 0, st>>>(w.Qb, w.Kb, w.Vb, w.Ob, types, LLEN, use_mask, Sq);
  launch_gemm(st, w.Ob, p.wo, p.bo, w.P1, M, DD, DD, 0);
  k_ln<<<dim3(M), dim3(256), 0, st>>>(w.P1, A, p.g1, p.c1, npm, LLEN, off, w.Ob, Sq);
  launch_gemm(st, w.Ob, p.w1, p.b1, w.Hb, M, DIH, DD, EPI_GELU);
  launch_gemm(st, w.Hb, p.w2, p.b2, w.P1, M, DD, DIH, 0);
  k_ln<<<dim3(M), dim3(256), 0, st>>>(w.P1, w.Ob, p.g2, p.c2, npm, LLEN, off, A, Sq);
}

extern "C" void kernel_launch(void* const* d_in, const int* in_sizes, int n_in,
                              void* d_out, int out_size, void* d_ws, size_t ws_size,
                              hipStream_t stream)
{
  (void)in_sizes; (void)n_in; (void)ws_size;
  // ----- inputs (setup_inputs order; params pytree flattened alphabetically) --
  const float* xy_pred    = (const float*)d_in[1];
  const float* event_time = (const float*)d_in[2];
  const float* attn_stack = (const float*)d_in[5];
  const float* enc_stack  = (const float*)d_in[6];
  const float* npm        = (const float*)d_in[7];   // non_pad_mask
  const float* npm2       = (const float*)d_in[8];   // non_pad_mask2
  const int pbase = 9;
  const float* emb = (const float*)d_in[pbase];
  LayerP lp[NLAY];
  for (int i = 0; i < NLAY; ++i) {
    int b0 = pbase + 1 + i * 16; // alphabetical leaf order within each layer dict
    lp[i].b1 = (const float*)d_in[b0 + 0];  lp[i].b2 = (const float*)d_in[b0 + 1];
    lp[i].bk = (const float*)d_in[b0 + 2];  lp[i].bo = (const float*)d_in[b0 + 3];
    lp[i].bq = (const float*)d_in[b0 + 4];  lp[i].bv = (const float*)d_in[b0 + 5];
    lp[i].c1 = (const float*)d_in[b0 + 6];  lp[i].c2 = (const float*)d_in[b0 + 7];
    lp[i].g1 = (const float*)d_in[b0 + 8];  lp[i].g2 = (const float*)d_in[b0 + 9];
    lp[i].w1 = (const float*)d_in[b0 + 10]; lp[i].w2 = (const float*)d_in[b0 + 11];
    lp[i].wk = (const float*)d_in[b0 + 12]; lp[i].wo = (const float*)d_in[b0 + 13];
    lp[i].wq = (const float*)d_in[b0 + 14]; lp[i].wv = (const float*)d_in[b0 + 15];
  }
  const int* etp = (const int*)d_in[pbase + 65];  // event_type_pred
  const int* et  = (const int*)d_in[pbase + 66];  // event_type

  // ----- workspace layout -----
  const size_t NBL = (size_t)BNUM * LLEN * DD;   // 2,097,152 floats
  const size_t NDI = (size_t)BNUM * LLEN * DIH;
  const size_t NY  = (size_t)BNUM * SEGC * DD;
  float* ws = (float*)d_ws;
  size_t off = 0;
  auto alloc = [&](size_t n) { float* p = ws + off; off += (n + 63) & ~(size_t)63; return p; };
  float* TEM = alloc(NBL);      // reused as OP for stack 3
  float* X   = alloc(NBL);
  Ctx cx;
  cx.Qb = alloc(NBL); cx.Kb = alloc(NBL); cx.Vb = alloc(NBL);
  cx.Ob = alloc(NBL); cx.P1 = alloc(NBL); cx.Hb = alloc(NDI);
  float* Yb  = alloc(NY);
  float* SPb = alloc(NY);
  float* ATT = alloc((size_t)HALFC * HALFC);
  float* G   = alloc(128);
  int*   IND = (int*)alloc(16);

  const int EW = 256;
  auto gsz = [&](size_t n) { return dim3((unsigned)((n + EW - 1) / EW)); };

  // ================= stack 1: encoder over event_type, length L, masked ======
  k_sincos<<<gsz(NBL), EW, 0, stream>>>(event_time, npm2, TEM, LLEN, LLEN, 0, (int)NBL);
  k_embed <<<gsz(NBL), EW, 0, stream>>>(et, emb, X, LLEN, LLEN, 0, (int)NBL);
  for (int l = 0; l < NLAY; ++l) {
    k_add<<<gsz(NBL), EW, 0, stream>>>(X, TEM, (int)NBL);
    run_layer(stream, X, LLEN, 1, etp, npm2, 0, lp[l], cx);
  }

  // ================= stack 2: decoder over last SEG, unmasked ================
  k_sincos<<<gsz(NY), EW, 0, stream>>>(xy_pred, npm, SPb, SEGC, LLEN, LLEN - SEGC, (int)NY);
  k_embed <<<gsz(NY), EW, 0, stream>>>(etp, emb, Yb, SEGC, LLEN, LLEN - SEGC, (int)NY);
  for (int l = 0; l < NLAY; ++l) {
    k_add<<<gsz(NY), EW, 0, stream>>>(Yb, SPb, (int)NY);
    run_layer(stream, Yb, SEGC, 0, etp, npm, LLEN - SEGC, lp[l], cx);
  }
  // attention probe from last layer's q,k (b=0, h=H-1), then retrieval
  k_attblock<<<dim3(HALFC), dim3(256), 0, stream>>>(cx.Qb, cx.Kb, ATT);
  k_zero<<<dim3(1), dim3(128), 0, stream>>>(G, NTC * NTC);
  k_graph<<<gsz((size_t)HALFC * HALFC), EW, 0, stream>>>(ATT, etp, LLEN, G);
  k_retrieve<<<dim3(1), dim3(128), 0, stream>>>(G, attn_stack, IND);
  k_concat<<<gsz(NBL), EW, 0, stream>>>(X, enc_stack, IND, TEM, (int)NBL);

  // ================= stack 3: encoder over concat, masked, no tem ===========
  for (int l = 0; l < NLAY; ++l)
    run_layer(stream, TEM, LLEN, 1, etp, npm2, 0, lp[l], cx);

  hipMemcpyAsync(d_out, TEM, NBL * sizeof(float), hipMemcpyDeviceToDevice, stream);
  (void)out_size;
}
